// GGNNRelReason_1529008357654
// MI455X (gfx1250) — matrix operand order repaired and checked by reading, not compiled, not run on hardware
//
#include <hip/hip_runtime.h>
#include <hip/hip_bf16.h>
#include <math.h>

// ---------------------------------------------------------------------------
// GGNN relation reasoning for MI455X (gfx1250, wave32).
// GEMM path: v_wmma_f32_16x16x32_bf16 (fp32 accum, bf16 operands), with the
// shared B panel staged into LDS via GLOBAL_LOAD_ASYNC_TO_LDS_B128 (ASYNCcnt),
// double-buffered; per K-tile all 4 B fragments are loaded from LDS up front
// so the 4 WMMAs issue back-to-back.
// ---------------------------------------------------------------------------

#define B_      8
#define OPI_    64
#define RPI_    1024
#define N_OBJ   (B_ * OPI_)      // 512
#define R_REL   (B_ * RPI_)      // 8192
#define OBJ_DIM 4096
#define H_      512
#define T_      3
#define NOC_    151
#define NRC_    51
#define NOC_PADK 160             // K-pad for probs/W_emb (mult of 32)
#define NOC_PADN 192             // N-pad for W_cls_obj (mult of 64)
#define NRC_PADN 64              // N-pad for W_cls_rel (mult of 64)
#define IOU_THR 0.3f

typedef unsigned short u16;
typedef __attribute__((ext_vector_type(8)))  __bf16 v8bf;
typedef __attribute__((ext_vector_type(16))) __bf16 v16bf;
typedef __attribute__((ext_vector_type(8)))  float  v8f;

__device__ __forceinline__ u16 f2bf(float f) {
  unsigned int u = __float_as_uint(f);
  u += 0x7FFFu + ((u >> 16) & 1u);   // round-to-nearest-even
  return (u16)(u >> 16);
}

// ---------------------------------------------------------------------------
// Utility kernels
// ---------------------------------------------------------------------------
__global__ void fill_f32(float* __restrict__ p, int n, float v) {
  int i = blockIdx.x * blockDim.x + threadIdx.x;
  if (i < n) p[i] = v;
}

__global__ void conv_bf16(const float* __restrict__ src, u16* __restrict__ dst, int n) {
  int i = blockIdx.x * blockDim.x + threadIdx.x;
  if (i < n) dst[i] = f2bf(src[i]);
}

// Pack a (K x Nv) fp32 weight slice (row stride srcld, column offset col0)
// into a WMMA-fragment-friendly bf16 layout:
//   dst[(kt*Np + n)*32 + kk]  where k = kt*32+kk.
// Zero-pads K -> Kp and Nv -> Np.
__global__ void pack_w(const float* __restrict__ src, int srcld, int col0,
                       int K, int Nv, int Kp, int Np, u16* __restrict__ dst) {
  int t = blockIdx.x * blockDim.x + threadIdx.x;
  int total = Kp * Np;
  if (t >= total) return;
  int kt = t / (Np * 32);
  int rem = t - kt * (Np * 32);
  int n  = rem >> 5;
  int kk = rem & 31;
  int k  = kt * 32 + kk;
  u16 v = 0;
  if (k < K && n < Nv) v = f2bf(src[(size_t)k * srcld + col0 + n]);
  dst[t] = v;
}

// Row softmax of obj_logits (N_OBJ x NOC_) -> padded bf16 (N_OBJ x NOC_PADK)
__global__ void softmax_pad_bf16(const float* __restrict__ logits, u16* __restrict__ probs) {
  int row = blockIdx.x * blockDim.x + threadIdx.x;
  if (row >= N_OBJ) return;
  const float* l = logits + (size_t)row * NOC_;
  float mx = l[0];
  for (int c = 1; c < NOC_; ++c) mx = fmaxf(mx, l[c]);
  float sum = 0.f;
  for (int c = 0; c < NOC_; ++c) sum += expf(l[c] - mx);
  float inv = 1.f / sum;
  u16* p = probs + (size_t)row * NOC_PADK;
  for (int c = 0; c < NOC_; ++c) p[c] = f2bf(expf(l[c] - mx) * inv);
  for (int c = NOC_; c < NOC_PADK; ++c) p[c] = 0;
}

// Row softmax into fp32 (for NMS scoring)
__global__ void softmax_f32(const float* __restrict__ logits, float* __restrict__ sp) {
  int row = blockIdx.x * blockDim.x + threadIdx.x;
  if (row >= N_OBJ) return;
  const float* l = logits + (size_t)row * NOC_;
  float mx = l[0];
  for (int c = 1; c < NOC_; ++c) mx = fmaxf(mx, l[c]);
  float sum = 0.f;
  for (int c = 0; c < NOC_; ++c) sum += expf(l[c] - mx);
  float inv = 1.f / sum;
  float* p = sp + (size_t)row * NOC_;
  for (int c = 0; c < NOC_; ++c) p[c] = expf(l[c] - mx) * inv;
}

// ---------------------------------------------------------------------------
// WMMA bf16 GEMM:  C[M x Nvalid] = A[M x Kp](bf16) @ Bp(packed bf16) (+bias)(+addC)
// Block = 128 threads (4 waves). Block tile = 64(M) x 64(N).
// Each wave owns a 16-row strip and 4 accumulators (16x64); the 64-col B panel
// per K-tile (4 KB) is staged to LDS with async-to-LDS copies, double-buffered.
// Requires: M % 64 == 0, Kp % 32 == 0, Np % 64 == 0, lda % 8 == 0.
// ---------------------------------------------------------------------------
#define PANEL_E 2048   // 64 cols * 32 kk  (u16 elements per K-tile panel)

__global__ void __launch_bounds__(128)
gemm_bf16_wmma(const u16* __restrict__ A, int lda,
               const u16* __restrict__ Bp, int Np,
               float* __restrict__ C, int ldc, int Nvalid, int Kp,
               const float* __restrict__ bias,
               const float* __restrict__ addC) {
  __shared__ __align__(16) u16 sB[2][PANEL_E];   // 8 KB double buffer

  const int lane  = threadIdx.x & 31;
  const int wave  = threadIdx.x >> 5;
  const int mBase = blockIdx.y * 64 + wave * 16;
  const int nBase = blockIdx.x * 64;
  const int l16   = lane & 15;
  const int kh    = lane >> 4;          // K-half selector (wave32 layout)
  const int mRow  = mBase + l16;        // A: row per lane

  v8f acc0 = {0.f,0.f,0.f,0.f,0.f,0.f,0.f,0.f};
  v8f acc1 = acc0, acc2 = acc0, acc3 = acc0;

  const int kTiles = Kp >> 5;
  const u16* aRow = A + (size_t)mRow * lda;

  // Stage the 4 KB B panel for k-tile kt into LDS buffer `buf`.
  // 128 threads x 2 x 16B async copies, tracked by ASYNCcnt.
  auto stage = [&](int kt, int buf) {
    const u16* src = Bp + ((size_t)kt * Np + nBase) * 32;
#pragma unroll
    for (int p = 0; p < 2; ++p) {
      int eoff = (p * 128 + threadIdx.x) * 8;              // 8 u16 = 16 B
      unsigned lds  = (unsigned)(unsigned long long)(&sB[buf][eoff]);
      unsigned long long ga = (unsigned long long)(src + eoff);
      asm volatile("global_load_async_to_lds_b128 %0, %1, off"
                   :: "v"(lds), "v"(ga) : "memory");
    }
  };

  stage(0, 0);
  for (int kt = 0; kt < kTiles; ++kt) {
    const int buf = kt & 1;
    const bool more = (kt + 1 < kTiles);

    // A fragment global loads issue first: latency overlaps the LDS sync.
    const u16* ap = aRow + kt * 32 + kh * 8;
    v8bf a_lo = *reinterpret_cast<const v8bf*>(ap);
    v8bf a_hi = *reinterpret_cast<const v8bf*>(ap + 16);
    v16bf a = __builtin_shufflevector(a_lo, a_hi, 0,1,2,3,4,5,6,7,8,9,10,11,12,13,14,15);
    __builtin_prefetch(ap + 128, 0, 0);   // global_prefetch_b8 for next A tiles

    if (more) {
      stage(kt + 1, buf ^ 1);
      asm volatile("s_wait_asynccnt 0x2" ::: "memory");  // current panel done
    } else {
      asm volatile("s_wait_asynccnt 0x0" ::: "memory");
    }
    __syncthreads();

    // Load ALL four B fragments from LDS first (clustered ds_load_b128, one
    // wait), so the 4 WMMAs can issue back-to-back on the matrix pipe.
    v16bf bfr[4];
#pragma unroll
    for (int nt = 0; nt < 4; ++nt) {
      const u16* bp = &sB[buf][(nt * 16 + l16) * 32 + kh * 8];
      v8bf b_lo = *reinterpret_cast<const v8bf*>(bp);
      v8bf b_hi = *reinterpret_cast<const v8bf*>(bp + 16);
      bfr[nt] = __builtin_shufflevector(b_lo, b_hi, 0,1,2,3,4,5,6,7,8,9,10,11,12,13,14,15);
    }
    acc0 = __builtin_amdgcn_wmma_f32_16x16x32_bf16(false, a, false, bfr[0], (short)0, acc0, false, false);
    acc1 = __builtin_amdgcn_wmma_f32_16x16x32_bf16(false, a, false, bfr[1], (short)0, acc1, false, false);
    acc2 = __builtin_amdgcn_wmma_f32_16x16x32_bf16(false, a, false, bfr[2], (short)0, acc2, false, false);
    acc3 = __builtin_amdgcn_wmma_f32_16x16x32_bf16(false, a, false, bfr[3], (short)0, acc3, false, false);
    __syncthreads();   // protect buffer before it is re-staged
  }

  // 32-bit C/D 16x16 layout: VGPR v -> M = v + 8*(lane>=16); N = lane&15
  const int mOff = kh * 8;
#pragma unroll
  for (int nt = 0; nt < 4; ++nt) {
    int nCol = nBase + nt * 16 + l16;
    if (nCol < Nvalid) {
      v8f accv = (nt == 0) ? acc0 : (nt == 1) ? acc1 : (nt == 2) ? acc2 : acc3;
      float bz = bias ? bias[nCol] : 0.f;
      for (int v = 0; v < 8; ++v) {
        int row = mBase + mOff + v;
        float val = accv[v] + bz;
        if (addC) val += addC[(size_t)row * ldc + nCol];
        C[(size_t)row * ldc + nCol] = val;
      }
    }
  }
}

// ---------------------------------------------------------------------------
// GGNN support kernels
// ---------------------------------------------------------------------------
__global__ void gather_so_bf16(const float* __restrict__ h_obj,
                               const int* __restrict__ rel_inds,
                               u16* __restrict__ so_bf) {
  int idx = blockIdx.x * blockDim.x + threadIdx.x;
  if (idx >= R_REL * H_) return;
  int r = idx / H_, j = idx - r * H_;
  int s = rel_inds[r * 3 + 1];
  int o = rel_inds[r * 3 + 2];
  so_bf[idx] = f2bf(h_obj[(size_t)s * H_ + j] + h_obj[(size_t)o * H_ + j]);
}

__global__ void scatter_mobj(const float* __restrict__ hm,
                             const int* __restrict__ rel_inds,
                             float* __restrict__ m_obj) {
  int idx = blockIdx.x * blockDim.x + threadIdx.x;
  if (idx >= R_REL * H_) return;
  int r = idx / H_, j = idx - r * H_;
  int s = rel_inds[r * 3 + 1];
  int o = rel_inds[r * 3 + 2];
  float v = hm[idx];
  atomicAdd(&m_obj[(size_t)s * H_ + j], v);
  atomicAdd(&m_obj[(size_t)o * H_ + j], v);
}

// z = sigmoid(xW[:, j] + hU[:, j] + b[j]); r = sigmoid(.. +H); rh = r*h (bf16)
__global__ void gru_zr(const float* __restrict__ xW, const float* __restrict__ hU,
                       const float* __restrict__ b, const float* __restrict__ h,
                       float* __restrict__ z, u16* __restrict__ rh_bf, int M) {
  int idx = blockIdx.x * blockDim.x + threadIdx.x;
  if (idx >= M * H_) return;
  int row = idx / H_, j = idx - row * H_;
  float az = xW[(size_t)row * (3 * H_) + j]      + hU[(size_t)row * (2 * H_) + j]      + b[j];
  float ar = xW[(size_t)row * (3 * H_) + H_ + j] + hU[(size_t)row * (2 * H_) + H_ + j] + b[H_ + j];
  float zz = 1.f / (1.f + expf(-az));
  float rr = 1.f / (1.f + expf(-ar));
  z[idx] = zz;
  rh_bf[idx] = f2bf(rr * h[idx]);
}

// n = tanh(xW[:, 2H+j] + hUn + b[2H+j]); h = (1-z)h + z n (writes f32 + bf16)
__global__ void gru_upd(const float* __restrict__ xW, const float* __restrict__ hUn,
                        const float* __restrict__ b, const float* __restrict__ z,
                        float* __restrict__ h, u16* __restrict__ h_bf, int M) {
  int idx = blockIdx.x * blockDim.x + threadIdx.x;
  if (idx >= M * H_) return;
  int row = idx / H_, j = idx - row * H_;
  float nn = tanhf(xW[(size_t)row * (3 * H_) + 2 * H_ + j] + hUn[idx] + b[2 * H_ + j]);
  float zz = z[idx];
  float hv = (1.f - zz) * h[idx] + zz * nn;
  h[idx] = hv;
  h_bf[idx] = f2bf(hv);
}

// ---------------------------------------------------------------------------
// Per-class NMS (class c = blockIdx.x + 1). 512 threads = 512 boxes.
// Stable rank-sort (matches jnp.argsort(-scores)) + greedy suppression.
// ---------------------------------------------------------------------------
__global__ void __launch_bounds__(512)
nms_kernel(const float* __restrict__ boxes_per_cls, const float* __restrict__ sp,
           float* __restrict__ keepmask) {
  const int c = blockIdx.x + 1;     // classes 1..150
  const int t = threadIdx.x;        // 0..511
  __shared__ float s_sc[N_OBJ];
  __shared__ float s_bx[N_OBJ][4];
  __shared__ float s_ar[N_OBJ];
  __shared__ int   s_ord[N_OBJ];
  __shared__ int   s_keep[N_OBJ];

  s_sc[t] = sp[(size_t)t * NOC_ + c];
  __syncthreads();
  float mys = s_sc[t];
  int rank = 0;
  for (int j = 0; j < N_OBJ; ++j) {
    float sj = s_sc[j];
    rank += (sj > mys) || (sj == mys && j < t);
  }
  s_ord[rank] = t;
  __syncthreads();

  int src = s_ord[t];
  const float* bb = boxes_per_cls + ((size_t)src * NOC_ + c) * 4;
  float x1 = bb[0], y1 = bb[1], x2 = bb[2], y2 = bb[3];
  s_bx[t][0] = x1; s_bx[t][1] = y1; s_bx[t][2] = x2; s_bx[t][3] = y2;
  s_ar[t] = (x2 - x1) * (y2 - y1);
  s_keep[t] = 1;

  for (int i = 0; i < N_OBJ; ++i) {
    __syncthreads();
    if (t > i && s_keep[i] && s_keep[t]) {
      float lx = fmaxf(s_bx[i][0], x1);
      float ly = fmaxf(s_bx[i][1], y1);
      float rx = fminf(s_bx[i][2], x2);
      float ry = fminf(s_bx[i][3], y2);
      float w  = fmaxf(rx - lx, 0.f);
      float h  = fmaxf(ry - ly, 0.f);
      float inter = w * h;
      float iou = inter / (s_ar[i] + s_ar[t] - inter + 1e-9f);
      if (iou > IOU_THR) s_keep[t] = 0;
    }
  }
  __syncthreads();
  keepmask[(size_t)src * NOC_ + c] = (float)s_keep[t];
}

__global__ void argmax_pred(const float* __restrict__ sp, const float* __restrict__ keepmask,
                            float* __restrict__ preds) {
  int i = blockIdx.x * blockDim.x + threadIdx.x;
  if (i >= N_OBJ) return;
  float best = keepmask[(size_t)i * NOC_ + 1] * sp[(size_t)i * NOC_ + 1];
  int bc = 1;
  for (int c = 2; c < NOC_; ++c) {
    float v = keepmask[(size_t)i * NOC_ + c] * sp[(size_t)i * NOC_ + c];
    if (v > best) { best = v; bc = c; }
  }
  preds[i] = (float)bc;
}

// ---------------------------------------------------------------------------
// Host launcher
// ---------------------------------------------------------------------------
extern "C" void kernel_launch(void* const* d_in, const int* in_sizes, int n_in,
                              void* d_out, int out_size, void* d_ws, size_t ws_size,
                              hipStream_t stream) {
  // ---- inputs (setup_inputs dict order) ----
  const float* obj_fmaps   = (const float*)d_in[1];   // (512, 4096)
  const float* obj_logits  = (const float*)d_in[2];   // (512, 151)
  const int*   rel_inds    = (const int*)  d_in[3];   // (8192, 3)
  const float* vr          = (const float*)d_in[4];   // (8192, 4096)
  const float* boxes       = (const float*)d_in[5];   // (512, 151, 4)
  const float* W_obj_proj  = (const float*)d_in[6];
  const float* b_obj_proj  = (const float*)d_in[7];
  const float* W_rel_proj  = (const float*)d_in[8];
  const float* b_rel_proj  = (const float*)d_in[9];
  const float* W_emb       = (const float*)d_in[10];
  const float* W_msg_rel   = (const float*)d_in[11];
  const float* W_msg_obj   = (const float*)d_in[12];
  const float* W_gru_obj   = (const float*)d_in[13];
  const float* U_gru_obj   = (const float*)d_in[14];
  const float* b_gru_obj   = (const float*)d_in[15];
  const float* W_gru_rel   = (const float*)d_in[16];
  const float* U_gru_rel   = (const float*)d_in[17];
  const float* b_gru_rel   = (const float*)d_in[18];
  const float* W_cls_rel   = (const float*)d_in[19];
  const float* b_cls_rel   = (const float*)d_in[20];
  const float* W_cls_obj   = (const float*)d_in[21];
  const float* b_cls_obj   = (const float*)d_in[22];

  float* out          = (float*)d_out;
  float* out_objlog   = out;                               // 512*151
  float* out_preds    = out + N_OBJ * NOC_;                // 512
  float* out_rellog   = out + N_OBJ * NOC_ + N_OBJ;        // 8192*51

  // ---- workspace bump allocator ----
  char*  base = (char*)d_ws;
  size_t off  = 0;
  auto alloc = [&](size_t bytes) -> void* {
    void* p = base + off;
    off = (off + bytes + 255) & ~(size_t)255;
    return p;
  };
  // bf16 buffers
  u16* fmaps_bf   = (u16*)alloc((size_t)N_OBJ * OBJ_DIM * 2);
  u16* vr_bf      = (u16*)alloc((size_t)R_REL * OBJ_DIM * 2);
  u16* probs_bf   = (u16*)alloc((size_t)N_OBJ * NOC_PADK * 2);
  u16* Wobj_p     = (u16*)alloc((size_t)OBJ_DIM * H_ * 2);
  u16* Wrel_p     = (u16*)alloc((size_t)OBJ_DIM * H_ * 2);
  u16* Wemb_p     = (u16*)alloc((size_t)NOC_PADK * H_ * 2);
  u16* Wmsgrel_p  = (u16*)alloc((size_t)H_ * H_ * 2);
  u16* Wmsgobj_p  = (u16*)alloc((size_t)H_ * H_ * 2);
  u16* Wgru_rel_p = (u16*)alloc((size_t)H_ * 3 * H_ * 2);
  u16* U12_rel_p  = (u16*)alloc((size_t)H_ * 2 * H_ * 2);
  u16* Un_rel_p   = (u16*)alloc((size_t)H_ * H_ * 2);
  u16* Wgru_obj_p = (u16*)alloc((size_t)H_ * 3 * H_ * 2);
  u16* U12_obj_p  = (u16*)alloc((size_t)H_ * 2 * H_ * 2);
  u16* Un_obj_p   = (u16*)alloc((size_t)H_ * H_ * 2);
  u16* Wclsrel_p  = (u16*)alloc((size_t)H_ * NRC_PADN * 2);
  u16* Wclsobj_p  = (u16*)alloc((size_t)H_ * NOC_PADN * 2);
  u16* h_rel_bf   = (u16*)alloc((size_t)R_REL * H_ * 2);
  u16* h_obj_bf   = (u16*)alloc((size_t)N_OBJ * H_ * 2);
  u16* so_bf      = (u16*)alloc((size_t)R_REL * H_ * 2);
  u16* mrel_bf    = (u16*)alloc((size_t)R_REL * H_ * 2);
  u16* mobj_bf    = (u16*)alloc((size_t)N_OBJ * H_ * 2);
  u16* rh_rel_bf  = (u16*)alloc((size_t)R_REL * H_ * 2);
  u16* rh_obj_bf  = (u16*)alloc((size_t)N_OBJ * H_ * 2);
  // f32 buffers
  float* fo       = (float*)alloc((size_t)N_OBJ * H_ * 4);
  float* h_rel    = (float*)alloc((size_t)R_REL * H_ * 4);
  float* h_obj    = (float*)alloc((size_t)N_OBJ * H_ * 4);
  float* m_rel    = (float*)alloc((size_t)R_REL * H_ * 4);
  float* hm       = (float*)alloc((size_t)R_REL * H_ * 4);
  float* m_obj    = (float*)alloc((size_t)N_OBJ * H_ * 4);
  float* xW_rel   = (float*)alloc((size_t)R_REL * 3 * H_ * 4);
  float* hU_rel   = (float*)alloc((size_t)R_REL * 2 * H_ * 4);
  float* hUn_rel  = (float*)alloc((size_t)R_REL * H_ * 4);
  float* z_rel    = (float*)alloc((size_t)R_REL * H_ * 4);
  float* xW_obj   = (float*)alloc((size_t)N_OBJ * 3 * H_ * 4);
  float* hU_obj   = (float*)alloc((size_t)N_OBJ * 2 * H_ * 4);
  float* hUn_obj  = (float*)alloc((size_t)N_OBJ * H_ * 4);
  float* z_obj    = (float*)alloc((size_t)N_OBJ * H_ * 4);
  float* sp       = (float*)alloc((size_t)N_OBJ * NOC_ * 4);
  float* keepmask = (float*)alloc((size_t)N_OBJ * NOC_ * 4);

  auto cdiv = [](int a, int b) { return (a + b - 1) / b; };
  auto conv = [&](const float* s, u16* d, int n) {
    conv_bf16<<<cdiv(n, 256), 256, 0, stream>>>(s, d, n);
  };
  auto pack = [&](const float* s, int srcld, int col0, int K, int Nv, int Kp, int Np, u16* d) {
    pack_w<<<cdiv(Kp * Np, 256), 256, 0, stream>>>(s, srcld, col0, K, Nv, Kp, Np, d);
  };
  auto gemm = [&](const u16* A, int lda, const u16* Bp, int Np, float* C, int ldc,
                  int Nvalid, int Kp, const float* bias, const float* addC, int M) {
    dim3 grid(Np / 64, M / 64);
    gemm_bf16_wmma<<<grid, 128, 0, stream>>>(A, lda, Bp, Np, C, ldc, Nvalid, Kp, bias, addC);
  };

  // ---- convert activations / pack weights ----
  conv(obj_fmaps, fmaps_bf, N_OBJ * OBJ_DIM);
  conv(vr, vr_bf, R_REL * OBJ_DIM);
  softmax_pad_bf16<<<cdiv(N_OBJ, 256), 256, 0, stream>>>(obj_logits, probs_bf);

  pack(W_obj_proj, H_,      0, OBJ_DIM, H_,      OBJ_DIM,  H_,       Wobj_p);
  pack(W_rel_proj, H_,      0, OBJ_DIM, H_,      OBJ_DIM,  H_,       Wrel_p);
  pack(W_emb,      H_,      0, NOC_,    H_,      NOC_PADK, H_,       Wemb_p);
  pack(W_msg_rel,  H_,      0, H_,      H_,      H_,       H_,       Wmsgrel_p);
  pack(W_msg_obj,  H_,      0, H_,      H_,      H_,       H_,       Wmsgobj_p);
  pack(W_gru_rel,  3 * H_,  0, H_,      3 * H_,  H_,       3 * H_,   Wgru_rel_p);
  pack(U_gru_rel,  3 * H_,  0, H_,      2 * H_,  H_,       2 * H_,   U12_rel_p);
  pack(U_gru_rel,  3 * H_,  2 * H_, H_, H_,      H_,       H_,       Un_rel_p);
  pack(W_gru_obj,  3 * H_,  0, H_,      3 * H_,  H_,       3 * H_,   Wgru_obj_p);
  pack(U_gru_obj,  3 * H_,  0, H_,      2 * H_,  H_,       2 * H_,   U12_obj_p);
  pack(U_gru_obj,  3 * H_,  2 * H_, H_, H_,      H_,       H_,       Un_obj_p);
  pack(W_cls_rel,  NRC_,    0, H_,      NRC_,    H_,       NRC_PADN, Wclsrel_p);
  pack(W_cls_obj,  NOC_,    0, H_,      NOC_,    H_,       NOC_PADN, Wclsobj_p);

  // ---- projections ----
  gemm(fmaps_bf, OBJ_DIM, Wobj_p, H_, fo,    H_, H_, OBJ_DIM, b_obj_proj, nullptr, N_OBJ);
  gemm(vr_bf,    OBJ_DIM, Wrel_p, H_, h_rel, H_, H_, OBJ_DIM, b_rel_proj, nullptr, R_REL);
  conv(h_rel, h_rel_bf, R_REL * H_);
  // h_obj = fo + probs @ W_emb
  gemm(probs_bf, NOC_PADK, Wemb_p, H_, h_obj, H_, H_, NOC_PADK, nullptr, fo, N_OBJ);
  conv(h_obj, h_obj_bf, N_OBJ * H_);

  // ---- T GGNN iterations ----
  const int RH = R_REL * H_, NH = N_OBJ * H_;
  for (int t = 0; t < T_; ++t) {
    gather_so_bf16<<<cdiv(RH, 256), 256, 0, stream>>>(h_obj, rel_inds, so_bf);
    gemm(so_bf,    H_, Wmsgrel_p, H_, m_rel, H_, H_, H_, nullptr, nullptr, R_REL);
    conv(m_rel, mrel_bf, RH);
    gemm(h_rel_bf, H_, Wmsgobj_p, H_, hm,    H_, H_, H_, nullptr, nullptr, R_REL);
    fill_f32<<<cdiv(NH, 256), 256, 0, stream>>>(m_obj, NH, 0.f);
    scatter_mobj<<<cdiv(RH, 256), 256, 0, stream>>>(hm, rel_inds, m_obj);
    conv(m_obj, mobj_bf, NH);

    // rel GRU (h_rel, x = m_rel)
    gemm(mrel_bf,   H_, Wgru_rel_p, 3 * H_, xW_rel, 3 * H_, 3 * H_, H_, nullptr, nullptr, R_REL);
    gemm(h_rel_bf,  H_, U12_rel_p,  2 * H_, hU_rel, 2 * H_, 2 * H_, H_, nullptr, nullptr, R_REL);
    gru_zr<<<cdiv(RH, 256), 256, 0, stream>>>(xW_rel, hU_rel, b_gru_rel, h_rel, z_rel, rh_rel_bf, R_REL);
    gemm(rh_rel_bf, H_, Un_rel_p,   H_,     hUn_rel, H_,    H_,     H_, nullptr, nullptr, R_REL);
    gru_upd<<<cdiv(RH, 256), 256, 0, stream>>>(xW_rel, hUn_rel, b_gru_rel, z_rel, h_rel, h_rel_bf, R_REL);

    // obj GRU (h_obj, x = m_obj)
    gemm(mobj_bf,   H_, Wgru_obj_p, 3 * H_, xW_obj, 3 * H_, 3 * H_, H_, nullptr, nullptr, N_OBJ);
    gemm(h_obj_bf,  H_, U12_obj_p,  2 * H_, hU_obj, 2 * H_, 2 * H_, H_, nullptr, nullptr, N_OBJ);
    gru_zr<<<cdiv(NH, 256), 256, 0, stream>>>(xW_obj, hU_obj, b_gru_obj, h_obj, z_obj, rh_obj_bf, N_OBJ);
    gemm(rh_obj_bf, H_, Un_obj_p,   H_,     hUn_obj, H_,    H_,     H_, nullptr, nullptr, N_OBJ);
    gru_upd<<<cdiv(NH, 256), 256, 0, stream>>>(xW_obj, hUn_obj, b_gru_obj, z_obj, h_obj, h_obj_bf, N_OBJ);
  }

  // ---- classifiers (write straight into d_out) ----
  gemm(h_rel_bf, H_, Wclsrel_p, NRC_PADN, out_rellog, NRC_, NRC_, H_, b_cls_rel, nullptr, R_REL);
  gemm(h_obj_bf, H_, Wclsobj_p, NOC_PADN, out_objlog, NOC_, NOC_, H_, b_cls_obj, nullptr, N_OBJ);

  // ---- NMS + predictions ----
  softmax_f32<<<cdiv(N_OBJ, 256), 256, 0, stream>>>(out_objlog, sp);
  fill_f32<<<cdiv(N_OBJ * NOC_, 256), 256, 0, stream>>>(keepmask, N_OBJ * NOC_, 0.f);
  nms_kernel<<<NOC_ - 1, N_OBJ, 0, stream>>>(boxes, sp, keepmask);
  argmax_pred<<<cdiv(N_OBJ, 256), 256, 0, stream>>>(sp, keepmask, out_preds);
}
